// Sea_Attention_54279796687144
// MI455X (gfx1250) — matrix operand order repaired and checked by reading, not compiled, not run on hardware
//
#include <hip/hip_runtime.h>

// -------- CDNA5 (gfx1250) wave32 WMMA types --------
typedef __attribute__((ext_vector_type(16))) __bf16   v16bf;
typedef __attribute__((ext_vector_type(8)))  float    v8f;
typedef __attribute__((ext_vector_type(4)))  unsigned v4u;

static __device__ __forceinline__ unsigned short f2bfu(float f) {
  unsigned u = __builtin_bit_cast(unsigned, f);
  return (unsigned short)((u + 0x7FFFu + ((u >> 16) & 1u)) >> 16); // RNE
}
static __device__ __forceinline__ __bf16 f2bf(float f) {
  unsigned short h = f2bfu(f);
  return __builtin_bit_cast(__bf16, h);
}

// =====================================================================
// GEMM with BN-fold epilogue: Y[o,n] = f((sum_c W[o,c]*X[c,n])*scl[o]+bia[o])
// mode 0: store; mode 1: Y = |Y_prev - y|; mode 2: h_sigmoid
// Block: 256 thr = 8 waves (2x4). Block tile 32(O) x 128(N), TILE_K = 64.
// Wave tile 16x32 -> 4 x v_wmma_f32_16x16x32_bf16 per K step.
// LDS: A row-major bf16 [32][72], B transposed bf16 Bt[n][k] [128][72];
// both layouts make every fragment two contiguous ds_load_b128 per lane.
// =====================================================================
#define ASTR 72
#define BSTR 72

__global__ __launch_bounds__(256)
void k_gemm_bn(const float* __restrict__ Wt, const __bf16* __restrict__ X,
               const float* __restrict__ scl, const float* __restrict__ bia,
               float* __restrict__ Y,
               int O, int C, int N, long Xbs, long Ybs, int mode)
{
  __shared__ unsigned short As[32 * ASTR];
  __shared__ unsigned short Bt[128 * BSTR];

  const unsigned short* Xs = (const unsigned short*)X + (long)blockIdx.z * Xbs;
  Y += (long)blockIdx.z * Ybs;
  const int o0 = blockIdx.y * 32;
  const int n0 = blockIdx.x * 128;
  const int tid  = threadIdx.x;
  const int wave = tid >> 5, lane = tid & 31;
  const int wm = wave >> 2, wn = wave & 3;     // 2 x 4 wave grid
  const int m  = lane & 15, hi = lane >> 4;

  v8f acc[2];
  acc[0] = (v8f){0.f,0.f,0.f,0.f,0.f,0.f,0.f,0.f};
  acc[1] = (v8f){0.f,0.f,0.f,0.f,0.f,0.f,0.f,0.f};

  for (int k0 = 0; k0 < C; k0 += 64) {
    // ---- stage A tile 32(O) x 64(K): float4 loads, pack 4 bf16 -> b64 store
#pragma unroll
    for (int t = 0; t < 2; ++t) {
      int e = tid + t * 256;
      int r = e >> 4, c4 = (e & 15) * 4;
      float4 w4 = *(const float4*)(Wt + (long)(o0 + r) * C + k0 + c4);
      uint2 p;
      p.x = (unsigned)f2bfu(w4.x) | ((unsigned)f2bfu(w4.y) << 16);
      p.y = (unsigned)f2bfu(w4.z) | ((unsigned)f2bfu(w4.w) << 16);
      *(uint2*)&As[r * ASTR + c4] = p;
    }
    // ---- stage B tile 64(K) x 128(N): uint4 loads (8 bf16), transpose into Bt[n][k]
#pragma unroll
    for (int t = 0; t < 4; ++t) {
      int e = tid + t * 256;
      int r = e >> 4;          // K row 0..63
      int nb = (e & 15) * 8;   // N base
      uint4 x4 = *(const uint4*)(Xs + (long)(k0 + r) * N + n0 + nb);
      Bt[(nb + 0) * BSTR + r] = (unsigned short)(x4.x);
      Bt[(nb + 1) * BSTR + r] = (unsigned short)(x4.x >> 16);
      Bt[(nb + 2) * BSTR + r] = (unsigned short)(x4.y);
      Bt[(nb + 3) * BSTR + r] = (unsigned short)(x4.y >> 16);
      Bt[(nb + 4) * BSTR + r] = (unsigned short)(x4.z);
      Bt[(nb + 5) * BSTR + r] = (unsigned short)(x4.z >> 16);
      Bt[(nb + 6) * BSTR + r] = (unsigned short)(x4.w);
      Bt[(nb + 7) * BSTR + r] = (unsigned short)(x4.w >> 16);
    }
    // gfx1250 prefetch of next K tile
    if (k0 + 64 < C)
      __builtin_prefetch(Xs + (long)(k0 + 64 + (tid >> 4)) * N + n0 + (tid & 15) * 8, 0, 0);
    __syncthreads();

#pragma unroll
    for (int kc = 0; kc < 2; ++kc) {
      // A fragment: lanes hi=0 take K {0..7,16..23}, hi=1 take {8..15,24..31}
      const unsigned short* arow = &As[(wm * 16 + m) * ASTR + kc * 32];
      v4u alo = *(const v4u*)(arow + hi * 8);
      v4u ahi = *(const v4u*)(arow + 16 + hi * 8);
      v16bf af = __builtin_bit_cast(
          v16bf, __builtin_shufflevector(alo, ahi, 0,1,2,3,4,5,6,7));
#pragma unroll
      for (int sub = 0; sub < 2; ++sub) {
        // B fragment: lanes hi=0 take K {0..7,16..23}, hi=1 take {8..15,24..31}
        const unsigned short* brow =
            &Bt[(wn * 32 + sub * 16 + m) * BSTR + kc * 32];
        v4u blo = *(const v4u*)(brow + hi * 8);
        v4u bhi = *(const v4u*)(brow + 16 + hi * 8);
        v16bf bfrag = __builtin_bit_cast(
            v16bf, __builtin_shufflevector(blo, bhi, 0,1,2,3,4,5,6,7));
        acc[sub] = __builtin_amdgcn_wmma_f32_16x16x32_bf16(
            false, af, false, bfrag, (short)0, acc[sub], false, false);
      }
    }
    __syncthreads();
  }

  // C/D layout: VGPR r -> M = r + 8*hi, N = lane&15
#pragma unroll
  for (int sub = 0; sub < 2; ++sub) {
#pragma unroll
    for (int r = 0; r < 8; ++r) {
      int o = o0 + wm * 16 + r + 8 * hi;
      int n = n0 + wn * 32 + sub * 16 + m;
      if (o < O && n < N) {
        float v   = acc[sub][r] * scl[o] + bia[o];
        long  off = (long)o * N + n;
        if (mode == 1)      v = fabsf(Y[off] - v);
        else if (mode == 2) v = fminf(fmaxf(v + 3.f, 0.f), 6.f) * (1.f / 6.f);
        Y[off] = v;
      }
    }
  }
}

// f32 -> bf16 bulk convert, 4 elems/thread (b128 in, b64 out)
__global__ __launch_bounds__(256)
void k_cvt_bf16(const float4* __restrict__ x, uint2* __restrict__ y, long n4)
{
  long i = (long)blockIdx.x * blockDim.x + threadIdx.x;
  if (i >= n4) return;
  float4 v = x[i];
  uint2 p;
  p.x = (unsigned)f2bfu(v.x) | ((unsigned)f2bfu(v.y) << 16);
  p.y = (unsigned)f2bfu(v.z) | ((unsigned)f2bfu(v.w) << 16);
  y[i] = p;
}

// 3x3 depthwise conv + BN + relu on virtual concat [q(128);k(128);v(256)],
// writes bf16 directly (input to pw GEMM).
__global__ __launch_bounds__(256)
void k_dw3x3(const float* __restrict__ q, const float* __restrict__ k,
             const float* __restrict__ v, const float* __restrict__ w9,
             const float* __restrict__ s, const float* __restrict__ b,
             __bf16* __restrict__ out, int B_, int H, int W_)
{
  long idx = (long)blockIdx.x * blockDim.x + threadIdx.x;
  long total = (long)B_ * 512 * H * W_;
  if (idx >= total) return;
  int x = (int)(idx % W_); long t = idx / W_;
  int y = (int)(t % H);    t /= H;
  int ch = (int)(t % 512); int bb = (int)(t / 512);
  const float* src;
  if (ch < 128)      src = q + ((long)bb * 128 +  ch       ) * (long)H * W_;
  else if (ch < 256) src = k + ((long)bb * 128 + (ch - 128)) * (long)H * W_;
  else               src = v + ((long)bb * 256 + (ch - 256)) * (long)H * W_;
  const float* wc = w9 + ch * 9;
  float acc = 0.f;
#pragma unroll
  for (int dy = -1; dy <= 1; ++dy) {
    int yy = y + dy;
    if (yy < 0 || yy >= H) continue;
#pragma unroll
    for (int dx = -1; dx <= 1; ++dx) {
      int xx = x + dx;
      if (xx < 0 || xx >= W_) continue;
      acc += wc[(dy + 1) * 3 + (dx + 1)] * src[(long)yy * W_ + xx];
    }
  }
  out[idx] = f2bf(fmaxf(acc * s[ch] + b[ch], 0.f));
}

// Shunting: outw[b,u,c,h] = mean_w4 x[b,c,h,w4*4+u]  (sh_w)
//           outh[b,u,c,w] = mean_h4 x[b,c,u*(H/4)+h4,w]  (sh_h)   (H==W==128)
__global__ __launch_bounds__(256)
void k_shunt(const float* __restrict__ x, float* __restrict__ outw,
             float* __restrict__ outh, int B_, int C, int H, int W_)
{
  long idx = (long)blockIdx.x * blockDim.x + threadIdx.x;
  long total = (long)B_ * 4 * C * H;
  if (idx >= total) return;
  int l = (int)(idx % H); long t = idx / H;
  int c = (int)(t % C);   t /= C;
  int u = (int)(t % 4);   int bb = (int)(t / 4);
  const float* xb = x + ((long)bb * C + c) * (long)H * W_;
  float sw = 0.f;
  for (int w4 = 0; w4 < W_ / 4; ++w4) sw += xb[(long)l * W_ + w4 * 4 + u];
  outw[idx] = sw * (4.0f / W_);
  float sh = 0.f;
  for (int h4 = 0; h4 < H / 4; ++h4) sh += xb[(long)(u * (H / 4) + h4) * W_ + l];
  outh[idx] = sh * (4.0f / H);
}

// Add linearly-interpolated positional embedding (align_corners=False)
__global__ __launch_bounds__(256)
void k_add_pe(float* __restrict__ sh, const float* __restrict__ pe,
              int B_, int C, int L, int S)
{
  long idx = (long)blockIdx.x * blockDim.x + threadIdx.x;
  long total = (long)B_ * 4 * C * L;
  if (idx >= total) return;
  int l = (int)(idx % L); long t = idx / L;
  int c = (int)(t % C);
  int u = (int)((t / C) % 4);
  float xf = fmaxf((l + 0.5f) * ((float)S / L) - 0.5f, 0.f);
  int i0 = (int)xf;
  int i1 = (i0 + 1 < S) ? i0 + 1 : S - 1;
  float w = xf - (float)i0;
  const float* p = pe + ((long)u * C + c) * S;
  sh[idx] += p[i0] * (1.f - w) + p[i1] * w;
}

// Axial attention for one (b, chunk, head): streaming softmax over L=128,
// folds both v1 and v2; writes relu'd bf16 in GEMM-ready layout
// fold[b][h*32+d][u*128 + i]   (fold is [B,256,512])
__global__ __launch_bounds__(128)
void k_attn(const float* __restrict__ qs, const float* __restrict__ ks,
            const float* __restrict__ v1s, const float* __restrict__ v2s,
            __bf16* __restrict__ fold1, __bf16* __restrict__ fold2, int B_)
{
  const int L = 128;
  __shared__ float ksh [16 * 128];
  __shared__ float v1sh[32 * 128];
  __shared__ float v2sh[32 * 128];
  int blk = blockIdx.x;              // bb*32 + u*8 + h
  int h = blk & 7, u = (blk >> 3) & 3, bb = blk >> 5;
  const float* qp  = qs  + (((long)bb * 4 + u) * 128 + h * 16) * L;
  const float* kp  = ks  + (((long)bb * 4 + u) * 128 + h * 16) * L;
  const float* v1p = v1s + (((long)bb * 4 + u) * 256 + h * 32) * L;
  const float* v2p = v2s + (((long)bb * 4 + u) * 256 + h * 32) * L;
  for (int e = threadIdx.x; e < 16 * 128; e += 128) ksh[e] = kp[e];
  for (int e = threadIdx.x; e < 32 * 128; e += 128) { v1sh[e] = v1p[e]; v2sh[e] = v2p[e]; }
  __syncthreads();

  int i = threadIdx.x;               // query row
  float qreg[16];
#pragma unroll
  for (int d = 0; d < 16; ++d) qreg[d] = qp[d * L + i];
  const float scale = 0.25f;         // KD^-0.5, KD=16

  float mx = -1e30f, sum = 0.f;
  for (int j = 0; j < L; ++j) {
    float sd = 0.f;
#pragma unroll
    for (int d = 0; d < 16; ++d) sd += qreg[d] * ksh[d * 128 + j];
    sd *= scale;
    if (sd > mx) { sum *= __expf(mx - sd); mx = sd; }
    sum += __expf(sd - mx);
  }
  float inv = 1.f / sum;

  float acc1[32], acc2[32];
#pragma unroll
  for (int d = 0; d < 32; ++d) { acc1[d] = 0.f; acc2[d] = 0.f; }
  for (int j = 0; j < L; ++j) {
    float sd = 0.f;
#pragma unroll
    for (int d = 0; d < 16; ++d) sd += qreg[d] * ksh[d * 128 + j];
    float e = __expf(sd * scale - mx);
#pragma unroll
    for (int d = 0; d < 32; ++d) {
      acc1[d] += e * v1sh[d * 128 + j];
      acc2[d] += e * v2sh[d * 128 + j];
    }
  }
  long col = (long)u * 128 + i;
#pragma unroll
  for (int d = 0; d < 32; ++d) {
    long off = ((long)bb * 256 + h * 32 + d) * 512 + col;
    fold1[off] = f2bf(fmaxf(acc1[d] * inv, 0.f));
    fold2[off] = f2bf(fmaxf(acc2[d] * inv, 0.f));
  }
}

// xx = |r1-r2|(row) + |c1-c2|(col) + |v1-v2|; write relu(xx) as bf16 for proj GEMM.
// Index algebra from the reference reshape: p=hh/8, tr=(hh%8)*4 + ww/32, tc=ww%32
__global__ __launch_bounds__(256)
void k_combine(const float* __restrict__ r1, const float* __restrict__ r2,
               const float* __restrict__ c1, const float* __restrict__ c2,
               const float* __restrict__ v1, const float* __restrict__ v2,
               __bf16* __restrict__ xxbf, int B_)
{
  long idx = (long)blockIdx.x * blockDim.x + threadIdx.x;
  long total = (long)B_ * 256 * 128 * 128;
  if (idx >= total) return;
  int ww = (int)(idx % 128); long t = idx / 128;
  int hh = (int)(t % 128);   t /= 128;
  int ch = (int)(t % 256);   int bb = (int)(t / 256);
  int tc = ww & 31, a = ww >> 5;
  int p = hh >> 3, tr = (hh & 7) * 4 + a;
  long base = ((long)bb * 256 + ch) * 512;
  float rv = fabsf(r1[base + p * 32 + tr] - r2[base + p * 32 + tr]);
  float cv = fabsf(c1[base + p * 32 + tc] - c2[base + p * 32 + tc]);
  float vv = fabsf(v1[idx] - v2[idx]);
  xxbf[idx] = f2bf(fmaxf(vv + rv + cv, 0.f));
}

// out = att*qkv1 ; out+n = att*qkv2  (float4 streams)
__global__ __launch_bounds__(256)
void k_finalmul(const float4* __restrict__ att, const float4* __restrict__ qkv1,
                const float4* __restrict__ qkv2, float4* __restrict__ out, long n4)
{
  long i = (long)blockIdx.x * blockDim.x + threadIdx.x;
  if (i >= n4) return;
  float4 a = att[i], p = qkv1[i], q = qkv2[i], o1, o2;
  o1.x = a.x * p.x; o1.y = a.y * p.y; o1.z = a.z * p.z; o1.w = a.w * p.w;
  o2.x = a.x * q.x; o2.y = a.y * q.y; o2.z = a.z * q.z; o2.w = a.w * q.w;
  out[i]      = o1;
  out[n4 + i] = o2;
}

// =====================================================================
extern "C" void kernel_launch(void* const* d_in, const int* in_sizes, int n_in,
                              void* d_out, int out_size, void* d_ws, size_t ws_size,
                              hipStream_t stream)
{
  (void)in_sizes; (void)n_in; (void)out_size; (void)ws_size;
  const int B = 2, H = 128, W = 128, N = H * W;

  const float* x1 = (const float*)d_in[0];
  const float* x2 = (const float*)d_in[1];
#define PW(i) ((const float*)d_in[i])
  // param indices (setup_inputs dict order):
  // to_q_1:2-4 to_q_2:5-7 to_k_1:8-10 to_k_2:11-13 to_v_1:14-16 to_v_2:17-19
  // proj:20-22 per_1:23-25 per_2:26-28 pec_1:29-31 pec_2:32-34
  // dw_1:35-37 dw_2:38-40 pw_1:41-43 pw_2:44-46
  // pe_rowq:47 pe_rowk:48 pe_colq:49 pe_colk:50

  size_t off = 0;
  auto alloc = [&](size_t bytes) -> void* {
    void* p = (char*)d_ws + off;
    off = (off + bytes + 255) & ~(size_t)255;
    return p;
  };
  __bf16* x1bf  = (__bf16*)alloc((size_t)B * 256 * N * 2);
  __bf16* x2bf  = (__bf16*)alloc((size_t)B * 256 * N * 2);
  float*  qb    = (float*) alloc((size_t)B * 128 * N * 4);
  float*  kb    = (float*) alloc((size_t)B * 128 * N * 4);
  float*  v1b   = (float*) alloc((size_t)B * 256 * N * 4);
  float*  v2b   = (float*) alloc((size_t)B * 256 * N * 4);
  __bf16* dwbf  = (__bf16*)alloc((size_t)B * 512 * N * 2);
  float*  qkv1  = (float*) alloc((size_t)B * 256 * N * 4);
  float*  qkv2  = (float*) alloc((size_t)B * 256 * N * 4);
  float*  shq_r = (float*) alloc((size_t)B * 4 * 128 * 128 * 4);
  float*  shq_c = (float*) alloc((size_t)B * 4 * 128 * 128 * 4);
  float*  shk_r = (float*) alloc((size_t)B * 4 * 128 * 128 * 4);
  float*  shk_c = (float*) alloc((size_t)B * 4 * 128 * 128 * 4);
  float*  shv1_r= (float*) alloc((size_t)B * 4 * 256 * 128 * 4);
  float*  shv1_c= (float*) alloc((size_t)B * 4 * 256 * 128 * 4);
  float*  shv2_r= (float*) alloc((size_t)B * 4 * 256 * 128 * 4);
  float*  shv2_c= (float*) alloc((size_t)B * 4 * 256 * 128 * 4);
  __bf16* f1r   = (__bf16*)alloc((size_t)B * 256 * 512 * 2);
  __bf16* f2r   = (__bf16*)alloc((size_t)B * 256 * 512 * 2);
  __bf16* f1c   = (__bf16*)alloc((size_t)B * 256 * 512 * 2);
  __bf16* f2c   = (__bf16*)alloc((size_t)B * 256 * 512 * 2);
  float*  rr1   = (float*) alloc((size_t)B * 256 * 512 * 4);
  float*  rr2   = (float*) alloc((size_t)B * 256 * 512 * 4);
  float*  cc1   = (float*) alloc((size_t)B * 256 * 512 * 4);
  float*  cc2   = (float*) alloc((size_t)B * 256 * 512 * 4);
  __bf16* xxbf  = (__bf16*)alloc((size_t)B * 256 * N * 2);
  float*  attb  = (float*) alloc((size_t)B * 256 * N * 4);

  const long nX  = (long)B * 256 * N;
  const long nX4 = nX / 4;
  const int  TH  = 256;
  k_cvt_bf16<<<dim3((unsigned)((nX4 + TH - 1) / TH)), TH, 0, stream>>>((const float4*)x1, (uint2*)x1bf, nX4);
  k_cvt_bf16<<<dim3((unsigned)((nX4 + TH - 1) / TH)), TH, 0, stream>>>((const float4*)x2, (uint2*)x2bf, nX4);

  // q / k / v projections (1x1 conv as WMMA GEMM); N tile = 128
  dim3 gq(N / 128,  64 / 32, B);
  dim3 gk(N / 128, 128 / 32, B);
  dim3 gv(N / 128, 256 / 32, B);
  k_gemm_bn<<<gq, 256, 0, stream>>>(PW(2),  x1bf, PW(3),  PW(4),  qb,               64, 256, N, (long)256*N, (long)128*N, 0);
  k_gemm_bn<<<gq, 256, 0, stream>>>(PW(5),  x2bf, PW(6),  PW(7),  qb + (long)64*N,  64, 256, N, (long)256*N, (long)128*N, 0);
  k_gemm_bn<<<gk, 256, 0, stream>>>(PW(8),  x1bf, PW(9),  PW(10), kb,              128, 256, N, (long)256*N, (long)128*N, 0);
  k_gemm_bn<<<gk, 256, 0, stream>>>(PW(11), x2bf, PW(12), PW(13), kb,              128, 256, N, (long)256*N, (long)128*N, 1);
  k_gemm_bn<<<gv, 256, 0, stream>>>(PW(14), x1bf, PW(15), PW(16), v1b,             256, 256, N, (long)256*N, (long)256*N, 0);
  k_gemm_bn<<<gv, 256, 0, stream>>>(PW(17), x2bf, PW(18), PW(19), v2b,             256, 256, N, (long)256*N, (long)256*N, 0);

  // dw + pw (qkv branches)
  const long ndw = (long)B * 512 * N;
  k_dw3x3<<<dim3((unsigned)((ndw + TH - 1) / TH)), TH, 0, stream>>>(qb, kb, v1b, PW(35), PW(36), PW(37), dwbf, B, H, W);
  k_gemm_bn<<<gv, 256, 0, stream>>>(PW(41), dwbf, PW(42), PW(43), qkv1, 256, 512, N, (long)512*N, (long)256*N, 0);
  k_dw3x3<<<dim3((unsigned)((ndw + TH - 1) / TH)), TH, 0, stream>>>(qb, kb, v2b, PW(38), PW(39), PW(40), dwbf, B, H, W);
  k_gemm_bn<<<gv, 256, 0, stream>>>(PW(44), dwbf, PW(45), PW(46), qkv2, 256, 512, N, (long)512*N, (long)256*N, 0);

  // shunting
  const long nshq = (long)B * 4 * 128 * 128;
  const long nshv = (long)B * 4 * 256 * 128;
  k_shunt<<<dim3((unsigned)((nshq + TH - 1) / TH)), TH, 0, stream>>>(qb,  shq_r,  shq_c,  B, 128, H, W);
  k_shunt<<<dim3((unsigned)((nshq + TH - 1) / TH)), TH, 0, stream>>>(kb,  shk_r,  shk_c,  B, 128, H, W);
  k_shunt<<<dim3((unsigned)((nshv + TH - 1) / TH)), TH, 0, stream>>>(v1b, shv1_r, shv1_c, B, 256, H, W);
  k_shunt<<<dim3((unsigned)((nshv + TH - 1) / TH)), TH, 0, stream>>>(v2b, shv2_r, shv2_c, B, 256, H, W);

  // positional embeddings (q/k only)
  k_add_pe<<<dim3((unsigned)((nshq + TH - 1) / TH)), TH, 0, stream>>>(shq_r, PW(47), B, 128, 128, 16);
  k_add_pe<<<dim3((unsigned)((nshq + TH - 1) / TH)), TH, 0, stream>>>(shk_r, PW(48), B, 128, 128, 16);
  k_add_pe<<<dim3((unsigned)((nshq + TH - 1) / TH)), TH, 0, stream>>>(shq_c, PW(49), B, 128, 128, 16);
  k_add_pe<<<dim3((unsigned)((nshq + TH - 1) / TH)), TH, 0, stream>>>(shk_c, PW(50), B, 128, 128, 16);

  // axial attention (row & col)
  k_attn<<<dim3((unsigned)(B * 32)), 128, 0, stream>>>(shq_r, shk_r, shv1_r, shv2_r, f1r, f2r, B);
  k_attn<<<dim3((unsigned)(B * 32)), 128, 0, stream>>>(shq_c, shk_c, shv1_c, shv2_c, f1c, f2c, B);

  // per_/pec_ GEMMs on fold outputs [256 x 512]
  dim3 gp(512 / 128, 256 / 32, B);
  k_gemm_bn<<<gp, 256, 0, stream>>>(PW(23), f1r, PW(24), PW(25), rr1, 256, 256, 512, (long)256*512, (long)256*512, 0);
  k_gemm_bn<<<gp, 256, 0, stream>>>(PW(26), f2r, PW(27), PW(28), rr2, 256, 256, 512, (long)256*512, (long)256*512, 0);
  k_gemm_bn<<<gp, 256, 0, stream>>>(PW(29), f1c, PW(30), PW(31), cc1, 256, 256, 512, (long)256*512, (long)256*512, 0);
  k_gemm_bn<<<gp, 256, 0, stream>>>(PW(32), f2c, PW(33), PW(34), cc2, 256, 256, 512, (long)256*512, (long)256*512, 0);

  // combine -> relu(xx) as bf16
  k_combine<<<dim3((unsigned)((nX + TH - 1) / TH)), TH, 0, stream>>>(rr1, rr2, cc1, cc2, v1b, v2b, xxbf, B);

  // proj + h_sigmoid
  k_gemm_bn<<<gv, 256, 0, stream>>>(PW(20), xxbf, PW(21), PW(22), attb, 256, 256, N, (long)256*N, (long)256*N, 2);

  // final elementwise products -> concatenated outputs
  k_finalmul<<<dim3((unsigned)((nX4 + TH - 1) / TH)), TH, 0, stream>>>((const float4*)attb, (const float4*)qkv1, (const float4*)qkv2, (float4*)d_out, nX4);
#undef PW
}
